// C2PSA_62878321213967
// MI455X (gfx1250) — matrix operand order repaired and checked by reading, not compiled, not run on hardware
//
#include <hip/hip_runtime.h>
#include <hip/hip_bf16.h>
#include <stdint.h>

typedef _Float16 f16;
typedef __attribute__((ext_vector_type(8)))  _Float16 v8h;
typedef __attribute__((ext_vector_type(16))) _Float16 v16h;
typedef __attribute__((ext_vector_type(8)))  float    v8f;

#define CIN  256
#define CR   64
#define NTOT 4096
#define BATCH 4
#define KPAD 72   // 32-key K rows padded: 144B stride, conflict-free, 16B aligned
#define VPAD 40   // V^T rows padded: 80B stride, conflict-free, 16B aligned

__device__ __forceinline__ v16h cat8(v8h lo, v8h hi) {
    return __builtin_shufflevector(lo, hi, 0,1,2,3,4,5,6,7,8,9,10,11,12,13,14,15);
}

__device__ __forceinline__ v8f wmma16(v16h a, v16h b, v8f c) {
    // D = A(16x32 f16) x B(32x16 f16) + C(16x16 f32)
    return __builtin_amdgcn_wmma_f32_16x16x32_f16(false, a, false, b, (short)0, c, false, false);
}

// One 16-byte async copy global -> LDS (ASYNCcnt-tracked, per-lane).
__device__ __forceinline__ void async_b128(uint32_t lds_addr, const f16* gaddr) {
    asm volatile("global_load_async_to_lds_b128 %0, %1, off"
                 :: "v"(lds_addr), "v"(gaddr) : "memory");
}

// ---------------------------------------------------------------------------
// Kernel 1: weights f32 -> f16 (Wq scaled by 1/sqrt(64)).
// ---------------------------------------------------------------------------
__global__ void convert_w_kernel(const float* __restrict__ Wq,
                                 const float* __restrict__ Wk,
                                 const float* __restrict__ Wv,
                                 const float* __restrict__ Wp,
                                 f16* __restrict__ wh) {
    int idx = blockIdx.x * 256 + threadIdx.x;           // 65536 total
    float v;
    if (idx < 16384)       v = Wq[idx] * 0.125f;
    else if (idx < 32768)  v = Wk[idx - 16384];
    else if (idx < 49152)  v = Wv[idx - 32768];
    else                   v = Wp[idx - 49152];
    wh[idx] = (f16)v;
}

// ---------------------------------------------------------------------------
// Kernel 2: qkv projection with WMMA (q,k row-major (B,N,64); v stored (B,64,N)).
// ---------------------------------------------------------------------------
__global__ __launch_bounds__(128) void qkv_kernel(
        const float* __restrict__ x,
        const f16* __restrict__ Wqh, const f16* __restrict__ Wkh,
        const f16* __restrict__ Wvh,
        f16* __restrict__ qh, f16* __restrict__ kh, f16* __restrict__ vth) {
    __shared__ __align__(16) f16 xT[64][CIN];           // 32 KB
    const int b     = blockIdx.y;
    const int nbase = blockIdx.x * 64;
    const int tid   = threadIdx.x;
    const float* xb = x + (size_t)b * CIN * NTOT;

#pragma unroll
    for (int it = 0; it < 32; ++it) {
        int idx = it * 128 + tid;                       // [0, 4096)
        int c  = idx >> 4;
        int i0 = (idx & 15) * 4;
        float4 xv = *(const float4*)(xb + (size_t)c * NTOT + nbase + i0);
        xT[i0 + 0][c] = (f16)xv.x;
        xT[i0 + 1][c] = (f16)xv.y;
        xT[i0 + 2][c] = (f16)xv.z;
        xT[i0 + 3][c] = (f16)xv.w;
    }
    __syncthreads();

    const int lane = tid & 31, w = tid >> 5;
    const int lm = lane & 15, hw = lane >> 4;
    const int n = nbase + w * 16 + lm;
    const f16* Ws[3] = {Wqh, Wkh, Wvh};

#pragma unroll
    for (int mat = 0; mat < 3; ++mat) {
        const f16* W = Ws[mat];
#pragma unroll
        for (int rt = 0; rt < 4; ++rt) {
            v8f acc = {};
#pragma unroll
            for (int cc = 0; cc < 8; ++cc) {
                const f16* wrow = W + (size_t)(rt * 16 + lm) * CIN + cc * 32;
                v8h alo = *(const v8h*)(wrow + hw * 16);
                v8h ahi = *(const v8h*)(wrow + hw * 16 + 8);
                v16h bfr = *(const v16h*)&xT[w * 16 + lm][cc * 32 + hw * 16];
                acc = wmma16(cat8(alo, ahi), bfr, acc);
            }
            if (mat < 2) {
                v8h ph;
#pragma unroll
                for (int i = 0; i < 8; ++i) ph[i] = (f16)acc[i];
                f16* dst = (mat == 0 ? qh : kh)
                         + ((size_t)b * NTOT + n) * CR + rt * 16 + hw * 8;
                *(v8h*)dst = ph;
            } else {
#pragma unroll
                for (int i = 0; i < 8; ++i)
                    vth[((size_t)b * CR + rt * 16 + hw * 8 + i) * NTOT + n] =
                        (f16)acc[i];
            }
        }
    }
}

// ---------------------------------------------------------------------------
// Kernel 3: flash attention + fused output projection + residual.
// K / V^T chunks are staged in LDS with double-buffered async copies.
// ---------------------------------------------------------------------------
__global__ __launch_bounds__(128) void attn_kernel(
        const float* __restrict__ x,
        const f16* __restrict__ qh, const f16* __restrict__ kh,
        const f16* __restrict__ vth, const f16* __restrict__ Wph,
        float* __restrict__ out) {
    __shared__ __align__(16) f16 Kl[2][32][KPAD];       // 2 x 4.5 KB
    __shared__ __align__(16) f16 Vl[2][CR][VPAD];       // 2 x 5 KB
    __shared__ __align__(16) f16 Plds[4][16][32];       // wave-private P^T
    __shared__ __align__(16) f16 outlds[64][CR];        // out (n x d)

    const int b     = blockIdx.y;
    const int nbase = blockIdx.x * 64;
    const int tid = threadIdx.x, lane = tid & 31, w = tid >> 5;
    const int lm = lane & 15, hw = lane >> 4;

    const f16* qb = qh  + (size_t)b * NTOT * CR;
    const f16* kb = kh  + (size_t)b * NTOT * CR;
    const f16* vb = vth + (size_t)b * CR * NTOT;
    const int nq = nbase + w * 16 + lm;

    // Per-thread async-copy slots (same every chunk).
    const int krow = tid >> 2, kseg = tid & 3;          // K: 32 rows x 8 segs
    const int vrow = tid >> 1, vseg = tid & 1;          // V: 64 rows x 4 segs
    const uint32_t kl0 = (uint32_t)(uintptr_t)&Kl[0][krow][0];
    const uint32_t vl0 = (uint32_t)(uintptr_t)&Vl[0][vrow][0];
    const uint32_t kbufstep = (uint32_t)(32 * KPAD * sizeof(f16));
    const uint32_t vbufstep = (uint32_t)(CR * VPAD * sizeof(f16));
    const f16* kg = kb + (size_t)krow * CR;
    const f16* vg = vb + (size_t)vrow * NTOT;

    // Q B-fragments (persistent).
    v16h qf0 = *(const v16h*)(qb + (size_t)nq * CR + hw * 16);
    v16h qf1 = *(const v16h*)(qb + (size_t)nq * CR + 32 + hw * 16);

    v8f acc0 = {}, acc1 = {}, acc2 = {}, acc3 = {};
    float mrun = -1e30f, lrun = 0.0f;

    // Prologue: issue chunk 0 into buffer 0.
    async_b128(kl0 + kseg * 16u,       kg + kseg * 8);
    async_b128(kl0 + (kseg + 4) * 16u, kg + (kseg + 4) * 8);
    async_b128(vl0 + vseg * 16u,       vg + vseg * 8);
    async_b128(vl0 + (vseg + 2) * 16u, vg + (vseg + 2) * 8);

    int buf = 0;
    for (int jb = 0; jb < NTOT; jb += 32) {
        asm volatile("s_wait_asynccnt 0x0" ::: "memory");
        __syncthreads();
        if (jb + 32 < NTOT) {                           // issue next chunk
            const int nb = buf ^ 1;
            const f16* kgn = kg + (size_t)(jb + 32) * CR;
            const f16* vgn = vg + jb + 32;
            async_b128(kl0 + nb * kbufstep + kseg * 16u,       kgn + kseg * 8);
            async_b128(kl0 + nb * kbufstep + (kseg + 4) * 16u, kgn + (kseg + 4) * 8);
            async_b128(vl0 + nb * vbufstep + vseg * 16u,       vgn + vseg * 8);
            async_b128(vl0 + nb * vbufstep + (vseg + 2) * 16u, vgn + (vseg + 2) * 8);
        }

        // ---- ST tiles from LDS K ----
        v8f s0, s1;
#pragma unroll
        for (int kt = 0; kt < 2; ++kt) {
            const f16* krw = Kl[buf][kt * 16 + lm];
            v8h a0lo = *(const v8h*)(krw + hw * 16);
            v8h a0hi = *(const v8h*)(krw + hw * 16 + 8);
            v8h a1lo = *(const v8h*)(krw + 32 + hw * 16);
            v8h a1hi = *(const v8h*)(krw + 40 + hw * 16);
            v8f c = {};
            c = wmma16(cat8(a0lo, a0hi), qf0, c);
            c = wmma16(cat8(a1lo, a1hi), qf1, c);
            if (kt == 0) s0 = c; else s1 = c;
        }
        // ---- online softmax (per query column) ----
        float tmax = s0[0];
#pragma unroll
        for (int i = 1; i < 8; ++i) tmax = fmaxf(tmax, s0[i]);
#pragma unroll
        for (int i = 0; i < 8; ++i) tmax = fmaxf(tmax, s1[i]);
        tmax = fmaxf(tmax, __shfl_xor(tmax, 16));
        float mnew = fmaxf(mrun, tmax);
        float corr = __expf(mrun - mnew);
        float tsum = 0.0f;
        v8h p0, p1;
#pragma unroll
        for (int i = 0; i < 8; ++i) {
            float e = __expf(s0[i] - mnew); tsum += e; p0[i] = (f16)e;
        }
#pragma unroll
        for (int i = 0; i < 8; ++i) {
            float e = __expf(s1[i] - mnew); tsum += e; p1[i] = (f16)e;
        }
        tsum += __shfl_xor(tsum, 16);
        lrun = lrun * corr + tsum;
        mrun = mnew;
#pragma unroll
        for (int i = 0; i < 8; ++i) {
            acc0[i] *= corr; acc1[i] *= corr; acc2[i] *= corr; acc3[i] *= corr;
        }
        // ---- P^T via wave-private LDS ----
        *(v8h*)&Plds[w][lm][hw * 8]      = p0;
        *(v8h*)&Plds[w][lm][16 + hw * 8] = p1;
        asm volatile("s_wait_dscnt 0x0" ::: "memory");
        v16h pb = *(const v16h*)&Plds[w][lm][hw * 16];
        // ---- outT += VT x PT from LDS V ----
#pragma unroll
        for (int dt = 0; dt < 4; ++dt) {
            const f16* vrw = Vl[buf][dt * 16 + lm];
            v8h vlo = *(const v8h*)(vrw + hw * 16);
            v8h vhi = *(const v8h*)(vrw + hw * 16 + 8);
            v16h va = cat8(vlo, vhi);
            if (dt == 0) acc0 = wmma16(va, pb, acc0);
            else if (dt == 1) acc1 = wmma16(va, pb, acc1);
            else if (dt == 2) acc2 = wmma16(va, pb, acc2);
            else acc3 = wmma16(va, pb, acc3);
        }
        buf ^= 1;
    }

    // ---- normalize, stage out (n x d) ----
    const float linv = 1.0f / lrun;
#pragma unroll
    for (int dt = 0; dt < 4; ++dt) {
        v8f* a = (dt == 0) ? &acc0 : (dt == 1) ? &acc1 : (dt == 2) ? &acc2 : &acc3;
        v8h oh;
#pragma unroll
        for (int i = 0; i < 8; ++i) oh[i] = (f16)((*a)[i] * linv);
        *(v8h*)&outlds[w * 16 + lm][dt * 16 + hw * 8] = oh;
    }
    __syncthreads();

    // ---- fused output projection + residual ----
    const float* xb = x + (size_t)b * CIN * NTOT;
    float* yb = out + (size_t)b * CIN * NTOT;
#pragma unroll 4
    for (int ct = 0; ct < 16; ++ct) {
        v8f acc = {};
#pragma unroll
        for (int cc = 0; cc < 2; ++cc) {
            const f16* wrow = Wph + (size_t)(ct * 16 + lm) * CR + cc * 32;
            v8h wlo = *(const v8h*)(wrow + hw * 16);
            v8h whi = *(const v8h*)(wrow + hw * 16 + 8);
            v16h ob = *(const v16h*)&outlds[w * 16 + lm][cc * 32 + hw * 16];
            acc = wmma16(cat8(wlo, whi), ob, acc);
        }
#pragma unroll
        for (int i = 0; i < 8; ++i) {
            int c = ct * 16 + hw * 8 + i;
            size_t gi = (size_t)c * NTOT + nq;
            yb[gi] = acc[i] + xb[gi];
        }
    }
}

// ---------------------------------------------------------------------------
extern "C" void kernel_launch(void* const* d_in, const int* in_sizes, int n_in,
                              void* d_out, int out_size, void* d_ws, size_t ws_size,
                              hipStream_t stream) {
    const float* x  = (const float*)d_in[0];
    const float* Wq = (const float*)d_in[1];
    const float* Wk = (const float*)d_in[2];
    const float* Wv = (const float*)d_in[3];
    const float* Wp = (const float*)d_in[4];

    f16* wsh  = (f16*)d_ws;
    f16* Wqh  = wsh;
    f16* Wkh  = wsh + 16384;
    f16* Wvh  = wsh + 32768;
    f16* Wph  = wsh + 49152;
    f16* qh   = wsh + 65536;
    f16* kh   = qh  + (size_t)BATCH * NTOT * CR;
    f16* vth  = kh  + (size_t)BATCH * NTOT * CR;

    convert_w_kernel<<<256, 256, 0, stream>>>(Wq, Wk, Wv, Wp, wsh);
    qkv_kernel<<<dim3(NTOT / 64, BATCH), 128, 0, stream>>>(
        x, Wqh, Wkh, Wvh, qh, kh, vth);
    attn_kernel<<<dim3(NTOT / 64, BATCH), 128, 0, stream>>>(
        x, qh, kh, vth, Wph, (float*)d_out);
}
